// ActorCriticRNN_41712722378962
// MI455X (gfx1250) — matrix operand order repaired and checked by reading, not compiled, not run on hardware
//
#include <hip/hip_runtime.h>
#include <math.h>

// ---------- types for WMMA ----------
typedef __attribute__((ext_vector_type(16))) __bf16 v16bf;
typedef __attribute__((ext_vector_type(8)))  float  v8f;
typedef __attribute__((ext_vector_type(4)))  unsigned int u32x4;

union FragBF {
    v16bf v;
    u32x4 u[2];
};

__device__ __forceinline__ unsigned short f2bf(float f) {
    unsigned int x = __float_as_uint(f);
    unsigned int r = x + 0x7FFFu + ((x >> 16) & 1u);   // round-to-nearest-even
    return (unsigned short)(r >> 16);
}

// async global->LDS copy of 16 bytes (GVS mode: mem = saddr + vaddr32)
__device__ __forceinline__ void async_g2l_b128(unsigned int lds_off,
                                               unsigned int g_off,
                                               unsigned long long sbase) {
    asm volatile("global_load_async_to_lds_b128 %0, %1, %2"
                 :: "v"(lds_off), "v"(g_off), "s"(sbase)
                 : "memory");
}
__device__ __forceinline__ void wait_async0() {
    asm volatile("s_wait_asynccnt 0x0" ::: "memory");
}
__device__ __forceinline__ unsigned int lds_off32(const void* p) {
    return (unsigned int)(unsigned long long)p;   // LDS aperture: low 32 bits = offset
}

#define MAPW 24
#define NTEAMS 2
#define MAXU 16
#define NRELIC 6
#define BB 32
#define SS 128
#define RNN_D 64
#define NSAMP (BB * SS)          // 4096
#define FEAT_D (21 * 21 * 64)    // 28224

// ---------------------------------------------------------------------------
// K0a: f32 -> bf16 flat convert (used for Wi)
// ---------------------------------------------------------------------------
__global__ __launch_bounds__(256) void k_f32_to_bf16(const float* __restrict__ in,
                                                     unsigned short* __restrict__ out, int n) {
    int i = blockIdx.x * 256 + threadIdx.x;
    if (i < n) out[i] = f2bf(in[i]);
}

// ---------------------------------------------------------------------------
// K0b: conv weight (2,2,CIN,COUT) f32 -> wT (COUT, 4*CIN) bf16
// ---------------------------------------------------------------------------
__global__ __launch_bounds__(256) void k_transpose_w(const float* __restrict__ w,
                                                     unsigned short* __restrict__ wT,
                                                     int K4, int COUT) {
    int i = blockIdx.x * 256 + threadIdx.x;
    if (i < K4 * COUT) {
        int n = i / K4, k = i % K4;
        wT[i] = f2bf(w[(size_t)k * COUT + n]);
    }
}

// ---------------------------------------------------------------------------
// K1: build per-sample 24x24x24 feature grid (bf16 out)
// channels: 0-3 tile_emb, 4-5 unit counts/16, 6-21 unit emb (team-major),
//           22 relic count, 23 reward
// ---------------------------------------------------------------------------
__global__ __launch_bounds__(256) void k_build_grid(
    const int* __restrict__ positions,            // (B,S,2,16,2)
    const unsigned char* __restrict__ units_mask, // (B,S,2,16) bool
    const int* __restrict__ tile_type,            // (B,S,24,24)
    const int* __restrict__ relic_positions,      // (B,S,6,2)
    const unsigned char* __restrict__ relic_mask, // (B,S,6) bool
    const float* __restrict__ reward,             // (B,S)
    const float* __restrict__ emb_tile,           // (3,4)
    const float* __restrict__ emb_unit,           // (16,8)
    unsigned short* __restrict__ grid)            // (N,24,24,24) bf16
{
    __shared__ alignas(16) float g[MAPW * MAPW * 24];
    const int bs = blockIdx.x;
    const int tid = threadIdx.x;
    for (int i = tid; i < MAPW * MAPW * 24; i += 256) g[i] = 0.f;
    __syncthreads();

    const float rew = reward[bs];
    for (int cell = tid; cell < MAPW * MAPW; cell += 256) {
        int tt = tile_type[(size_t)bs * (MAPW * MAPW) + cell];
        float* c = &g[cell * 24];
        c[0] = emb_tile[tt * 4 + 0];
        c[1] = emb_tile[tt * 4 + 1];
        c[2] = emb_tile[tt * 4 + 2];
        c[3] = emb_tile[tt * 4 + 3];
        c[23] = rew;
    }
    if (tid < NTEAMS * MAXU) {
        int team = tid >> 4, unit = tid & 15;
        int base = (bs * NTEAMS + team) * MAXU + unit;
        float mk = units_mask[base] ? 1.f : 0.f;
        int py = positions[base * 2 + 0];
        int px = positions[base * 2 + 1];
        float* c = &g[(py * MAPW + px) * 24];
        atomicAdd(&c[4 + team], mk * (1.f / (float)MAXU));
        #pragma unroll
        for (int e = 0; e < 8; ++e)
            atomicAdd(&c[6 + team * 8 + e], mk * emb_unit[unit * 8 + e]);
    }
    if (tid >= 32 && tid < 32 + NRELIC) {
        int ri = tid - 32;
        int base = bs * NRELIC + ri;
        float mk = relic_mask[base] ? 1.f : 0.f;
        int py = relic_positions[base * 2 + 0];
        int px = relic_positions[base * 2 + 1];
        atomicAdd(&g[(py * MAPW + px) * 24 + 22], mk);
    }
    __syncthreads();
    unsigned short* go = grid + (size_t)bs * (MAPW * MAPW * 24);
    for (int i = tid; i < MAPW * MAPW * 24; i += 256) go[i] = f2bf(g[i]);
}

// ---------------------------------------------------------------------------
// K2/3/4: 2x2 VALID conv + bias + relu via im2col WMMA bf16
// in:  (N, HIN, HIN, CIN)  bf16    out: (N, HO, HO, COUT) bf16
// wT:  (COUT, 4*CIN)       bf16    bias: (COUT) f32
// LDS staging via GLOBAL_LOAD_ASYNC_TO_LDS_B128.
// ---------------------------------------------------------------------------
template <int CIN, int COUT, int HIN>
__global__ __launch_bounds__(128) void k_conv2x2_wmma(
    const unsigned short* __restrict__ in,
    const unsigned short* __restrict__ wT,
    const float* __restrict__ bias,
    unsigned short* __restrict__ out)
{
    constexpr int HO = HIN - 1;
    constexpr int K  = 4 * CIN;
    constexpr int M  = HO * HO;
    constexpr int MT = (M + 15) / 16;
    constexpr int NT = COUT / 16;
    constexpr int KT = K / 32;

    __shared__ alignas(16) unsigned short sIn[HIN * HIN * CIN];
    __shared__ alignas(16) unsigned short sW[COUT * K];

    const int smp = blockIdx.x;
    const int tid = threadIdx.x;

    { // async-stage input tile + weights into LDS (16B chunks; counts /8 exact)
        const unsigned long long gin =
            (unsigned long long)(in + (size_t)smp * (HIN * HIN * CIN));
        for (int i = tid; i < HIN * HIN * CIN / 8; i += 128)
            async_g2l_b128(lds_off32(&sIn[i * 8]), (unsigned int)(i * 16), gin);
        const unsigned long long gw = (unsigned long long)wT;
        for (int i = tid; i < COUT * K / 8; i += 128)
            async_g2l_b128(lds_off32(&sW[i * 8]), (unsigned int)(i * 16), gw);
        wait_async0();
    }
    __syncthreads();

    const int wave = tid >> 5;
    const int lane = tid & 31;
    const int half = lane >> 4;
    const int lr   = lane & 15;

    for (int mt = wave; mt < MT; mt += 4) {
        int m = mt * 16 + lr;
        if (m >= M) m = M - 1;                 // clamp pad rows (stores masked)
        const int oy = m / HO, ox = m % HO;

        v8f acc[NT];
        #pragma unroll
        for (int nt = 0; nt < NT; ++nt)
            #pragma unroll
            for (int j = 0; j < 8; ++j) acc[nt][j] = 0.f;

        #pragma unroll
        for (int kt = 0; kt < KT; ++kt) {
            // A fragment: row m, K-runs [kt*32+8*half, +8) and [kt*32+16+8*half, +8)
            FragBF a;
            {
                int k0 = kt * 32 + half * 8;
                int p = k0 / CIN, c = k0 % CIN;
                int iy = oy + (p >> 1), ix = ox + (p & 1);
                a.u[0] = *(const u32x4*)&sIn[(iy * HIN + ix) * CIN + c];
            }
            {
                int k1 = kt * 32 + 16 + half * 8;
                int p = k1 / CIN, c = k1 % CIN;
                int iy = oy + (p >> 1), ix = ox + (p & 1);
                a.u[1] = *(const u32x4*)&sIn[(iy * HIN + ix) * CIN + c];
            }
            #pragma unroll
            for (int nt = 0; nt < NT; ++nt) {
                // B fragment: col lr of n-tile, K-runs [kt*32+8*half, +8) and [kt*32+16+8*half, +8)
                FragBF b;
                const unsigned short* wp = &sW[(nt * 16 + lr) * K + kt * 32 + half * 8];
                b.u[0] = *(const u32x4*)wp;
                b.u[1] = *(const u32x4*)(wp + 16);
                acc[nt] = __builtin_amdgcn_wmma_f32_16x16x32_bf16(
                    false, a.v, false, b.v, (short)0, acc[nt], false, false);
            }
        }
        // epilogue: D layout -> col = lr, row = j + 8*half
        #pragma unroll
        for (int nt = 0; nt < NT; ++nt) {
            int col = nt * 16 + lr;
            float bv = bias[col];
            #pragma unroll
            for (int j = 0; j < 8; ++j) {
                int mm = mt * 16 + j + 8 * half;
                if (mm < M) {
                    float v = acc[nt][j] + bv;
                    v = v > 0.f ? v : 0.f;     // relu
                    out[((size_t)smp * M + mm) * COUT + col] = f2bf(v);
                }
            }
        }
    }
}

// ---------------------------------------------------------------------------
// K5: ig = feat(4096x28224) @ Wi.T + bi -> f32 (4096x192)
// Wi stored (192, 28224) == B^T, so B fragment rows are contiguous.
// 64 blocks x 4 waves; block does 64 rows, wave does 16 rows x full N=192.
// Double-buffered Wi k-slices staged via GLOBAL_LOAD_ASYNC_TO_LDS_B128.
// ---------------------------------------------------------------------------
__global__ __launch_bounds__(128) void k_gemm_ig(
    const unsigned short* __restrict__ feat,  // (4096, 28224) bf16
    const unsigned short* __restrict__ WiB,   // (192, 28224) bf16
    const float* __restrict__ bi,             // (192)
    float* __restrict__ ig)                   // (4096, 192)
{
    __shared__ alignas(16) unsigned short sW[2][192 * 32];

    const int blk  = blockIdx.x;
    const int tid  = threadIdx.x;
    const int wave = tid >> 5;
    const int lane = tid & 31;
    const int half = lane >> 4;
    const int lr   = lane & 15;

    const int mrow = blk * 64 + wave * 16 + lr;
    const unsigned short* arow = feat + (size_t)mrow * FEAT_D;

    // per-lane async-copy addressing: slice = 192 rows x 64B = 768 x b128
    // lane handles i = tid + j*128 -> row = i/4, 16B-chunk = i%4 (loop-invariant)
    unsigned int ldsoff[2][6];
    unsigned int goff[6];
    #pragma unroll
    for (int j = 0; j < 6; ++j) {
        int i = tid + j * 128;
        int row = i >> 2, chunk = i & 3;
        goff[j] = (unsigned int)((row * FEAT_D + chunk * 8) * 2);
        ldsoff[0][j] = lds_off32(&sW[0][row * 32 + chunk * 8]);
        ldsoff[1][j] = lds_off32(&sW[1][row * 32 + chunk * 8]);
    }

    v8f acc[12];
    #pragma unroll
    for (int nt = 0; nt < 12; ++nt)
        #pragma unroll
        for (int j = 0; j < 8; ++j) acc[nt][j] = 0.f;

    // prologue: stage k-slice 0 into buffer 0
    {
        const unsigned long long sbase = (unsigned long long)WiB;
        #pragma unroll
        for (int j = 0; j < 6; ++j) async_g2l_b128(ldsoff[0][j], goff[j], sbase);
        wait_async0();
    }
    __syncthreads();

    for (int kt = 0; kt < FEAT_D / 32; ++kt) {
        const int cur = kt & 1;
        const int nxt = cur ^ 1;
        if (kt + 1 < FEAT_D / 32) {   // prefetch next slice into other buffer
            const unsigned long long sbase =
                (unsigned long long)WiB + (unsigned long long)(kt + 1) * 64ull;
            #pragma unroll
            for (int j = 0; j < 6; ++j) async_g2l_b128(ldsoff[nxt][j], goff[j], sbase);
        }

        FragBF a;
        a.u[0] = *(const u32x4*)(arow + kt * 32 + half * 8);
        a.u[1] = *(const u32x4*)(arow + kt * 32 + 16 + half * 8);
        __builtin_prefetch(arow + kt * 32 + 64, 0, 1);

        #pragma unroll
        for (int nt = 0; nt < 12; ++nt) {
            FragBF b;
            const unsigned short* wp = &sW[cur][(nt * 16 + lr) * 32 + half * 8];
            b.u[0] = *(const u32x4*)wp;
            b.u[1] = *(const u32x4*)(wp + 16);
            acc[nt] = __builtin_amdgcn_wmma_f32_16x16x32_bf16(
                false, a.v, false, b.v, (short)0, acc[nt], false, false);
        }
        wait_async0();       // own async stores to sW[nxt] landed in LDS
        __syncthreads();     // publish sW[nxt]; all reads of sW[cur] are done
    }

    #pragma unroll
    for (int nt = 0; nt < 12; ++nt) {
        int col = nt * 16 + lr;
        float bv = bi[col];
        #pragma unroll
        for (int j = 0; j < 8; ++j) {
            int m = blk * 64 + wave * 16 + j + 8 * half;
            ig[(size_t)m * 192 + col] = acc[nt][j] + bv;
        }
    }
}

// ---------------------------------------------------------------------------
// K6: GRU scan. One block, 256 threads. h (32x64) f32 in LDS.
// ---------------------------------------------------------------------------
__global__ __launch_bounds__(256) void k_gru_scan(
    const float* __restrict__ ig,   // (4096, 192), row = b*S + t
    const float* __restrict__ Wh,   // (192, 64)
    const float* __restrict__ bn,   // (64)
    const float* __restrict__ h0,   // (32, 64)
    float* __restrict__ outbuf,     // (4096, 64)
    float* __restrict__ hid_out)    // (32, 64)
{
    __shared__ float h[BB * RNN_D];
    const int tid = threadIdx.x;
    for (int i = tid; i < BB * RNN_D; i += 256) h[i] = h0[i];
    __syncthreads();

    for (int t = 0; t < SS; ++t) {
        float hnew[8];
        #pragma unroll
        for (int j = 0; j < 8; ++j) {
            int e = tid + j * 256;
            int i = e >> 6, d = e & 63;
            const float* hr_ = &h[i * RNN_D];
            const float* wr = &Wh[(size_t)d * RNN_D];
            const float* wz = &Wh[(size_t)(64 + d) * RNN_D];
            const float* wn = &Wh[(size_t)(128 + d) * RNN_D];
            float sr = 0.f, sz = 0.f, sn = 0.f;
            for (int k = 0; k < RNN_D; ++k) {
                float hv = hr_[k];
                sr += hv * wr[k];
                sz += hv * wz[k];
                sn += hv * wn[k];
            }
            const float* igrow = &ig[(size_t)(i * SS + t) * 192];
            float r = 1.f / (1.f + __expf(-(igrow[d] + sr)));
            float z = 1.f / (1.f + __expf(-(igrow[64 + d] + sz)));
            float n = tanhf(igrow[128 + d] + r * (sn + bn[d]));
            hnew[j] = (1.f - z) * n + z * h[e];
        }
        __syncthreads();
        #pragma unroll
        for (int j = 0; j < 8; ++j) {
            int e = tid + j * 256;
            int i = e >> 6, d = e & 63;
            h[e] = hnew[j];
            outbuf[(size_t)(i * SS + t) * RNN_D + d] = hnew[j];
        }
        __syncthreads();
    }
    for (int i = tid; i < BB * RNN_D; i += 256) hid_out[i] = h[i];
}

// ---------------------------------------------------------------------------
// K7: heads. One block per row r of out (4096 rows).
// ---------------------------------------------------------------------------
__global__ __launch_bounds__(128) void k_heads(
    const float* __restrict__ outbuf,
    const float* __restrict__ Wa1, const float* __restrict__ ba1,
    const float* __restrict__ Wa2, const float* __restrict__ ba2,
    const float* __restrict__ Wc1, const float* __restrict__ bc1,
    const float* __restrict__ Wc2, const float* __restrict__ bc2,
    float* __restrict__ logits, float* __restrict__ values)
{
    __shared__ float row[64], a[64], c[64];
    const int r = blockIdx.x;
    const int tid = threadIdx.x;
    if (tid < 64) row[tid] = outbuf[(size_t)r * 64 + tid];
    __syncthreads();
    if (tid < 64) {
        float s1 = 0.f, s2 = 0.f;
        for (int k = 0; k < 64; ++k) {
            float rv = row[k];
            s1 += rv * Wa1[k * 64 + tid];
            s2 += rv * Wc1[k * 64 + tid];
        }
        a[tid] = tanhf(s1 + ba1[tid]);
        c[tid] = tanhf(s2 + bc1[tid]);
    }
    __syncthreads();
    if (tid < 96) {
        float s = 0.f;
        for (int k = 0; k < 64; ++k) s += a[k] * Wa2[k * 96 + tid];
        logits[(size_t)r * 96 + tid] = s + ba2[tid];
    } else if (tid == 96) {
        float s = 0.f;
        for (int k = 0; k < 64; ++k) s += c[k] * Wc2[k];
        values[r] = s + bc2[0];
    }
}

// ---------------------------------------------------------------------------
// host launcher
// ---------------------------------------------------------------------------
extern "C" void kernel_launch(void* const* d_in, const int* in_sizes, int n_in,
                              void* d_out, int out_size, void* d_ws, size_t ws_size,
                              hipStream_t stream) {
    (void)in_sizes; (void)n_in; (void)out_size; (void)ws_size;

    const int* positions            = (const int*)d_in[0];
    const unsigned char* units_mask = (const unsigned char*)d_in[1];
    const int* tile_type            = (const int*)d_in[2];
    const int* relic_positions      = (const int*)d_in[3];
    const unsigned char* relic_mask = (const unsigned char*)d_in[4];
    const float* reward             = (const float*)d_in[5];
    const float* hidden             = (const float*)d_in[6];
    const float* emb_tile           = (const float*)d_in[7];
    const float* emb_unit           = (const float*)d_in[8];
    const float* w1 = (const float*)d_in[9];
    const float* b1 = (const float*)d_in[10];
    const float* w2 = (const float*)d_in[11];
    const float* b2 = (const float*)d_in[12];
    const float* w3 = (const float*)d_in[13];
    const float* b3 = (const float*)d_in[14];
    const float* Wi = (const float*)d_in[15];
    const float* Wh = (const float*)d_in[16];
    const float* bi = (const float*)d_in[17];
    const float* bn = (const float*)d_in[18];
    const float* Wa1 = (const float*)d_in[19];
    const float* ba1 = (const float*)d_in[20];
    const float* Wa2 = (const float*)d_in[21];
    const float* ba2 = (const float*)d_in[22];
    const float* Wc1 = (const float*)d_in[23];
    const float* bc1 = (const float*)d_in[24];
    const float* Wc2 = (const float*)d_in[25];
    const float* bc2 = (const float*)d_in[26];

    // ---- workspace layout (ping-pong the big activation regions) ----
    const size_t SZ_GRID = (size_t)NSAMP * 24 * 24 * 24 * 2;   // 113.2 MB
    const size_t SZ_ACT1 = (size_t)NSAMP * 23 * 23 * 16 * 2;   //  69.3 MB
    const size_t SZ_ACT2 = (size_t)NSAMP * 22 * 22 * 32 * 2;   // 126.9 MB
    const size_t SZ_FEAT = (size_t)NSAMP * FEAT_D * 2;         // 231.2 MB
    const size_t REG_A = (SZ_GRID > SZ_ACT2) ? SZ_GRID : SZ_ACT2;
    const size_t REG_B = (SZ_ACT1 > SZ_FEAT) ? SZ_ACT1 : SZ_FEAT;

    char* ws = (char*)d_ws;
    size_t off = 0;
    auto take = [&](size_t bytes) -> char* {
        char* p = ws + off;
        off = (off + bytes + 255) & ~(size_t)255;
        return p;
    };
    char* regA   = take(REG_A);
    char* regB   = take(REG_B);
    unsigned short* wiB  = (unsigned short*)take((size_t)192 * FEAT_D * 2);
    unsigned short* wT1  = (unsigned short*)take((size_t)16 * 96 * 2);
    unsigned short* wT2  = (unsigned short*)take((size_t)32 * 64 * 2);
    unsigned short* wT3  = (unsigned short*)take((size_t)64 * 128 * 2);
    float* ig     = (float*)take((size_t)NSAMP * 192 * 4);
    float* outbuf = (float*)take((size_t)NSAMP * RNN_D * 4);

    unsigned short* grid_b = (unsigned short*)regA;   // then reused for act2
    unsigned short* act1   = (unsigned short*)regB;   // then reused for feat
    unsigned short* act2   = (unsigned short*)regA;
    unsigned short* feat   = (unsigned short*)regB;

    float* out_logits = (float*)d_out;                       // 4096*96
    float* out_values = out_logits + (size_t)NSAMP * 96;     // 4096
    float* out_hidden = out_values + NSAMP;                  // 2048

    // ---- weight conversions ----
    {
        int n = 192 * FEAT_D;
        k_f32_to_bf16<<<(n + 255) / 256, 256, 0, stream>>>(Wi, wiB, n);
    }
    k_transpose_w<<<(16 * 96 + 255) / 256, 256, 0, stream>>>(w1, wT1, 96, 16);
    k_transpose_w<<<(32 * 64 + 255) / 256, 256, 0, stream>>>(w2, wT2, 64, 32);
    k_transpose_w<<<(64 * 128 + 255) / 256, 256, 0, stream>>>(w3, wT3, 128, 64);

    // ---- pipeline ----
    k_build_grid<<<NSAMP, 256, 0, stream>>>(positions, units_mask, tile_type,
                                            relic_positions, relic_mask, reward,
                                            emb_tile, emb_unit, grid_b);

    k_conv2x2_wmma<24, 16, 24><<<NSAMP, 128, 0, stream>>>(grid_b, wT1, b1, act1);
    k_conv2x2_wmma<16, 32, 23><<<NSAMP, 128, 0, stream>>>(act1, wT2, b2, act2);
    k_conv2x2_wmma<32, 64, 22><<<NSAMP, 128, 0, stream>>>(act2, wT3, b3, feat);

    k_gemm_ig<<<NSAMP / 64, 128, 0, stream>>>(feat, wiB, bi, ig);

    k_gru_scan<<<1, 256, 0, stream>>>(ig, Wh, bn, hidden, outbuf, out_hidden);

    k_heads<<<NSAMP, 128, 0, stream>>>(outbuf, Wa1, ba1, Wa2, ba2,
                                       Wc1, bc1, Wc2, bc2,
                                       out_logits, out_values);
}
